// Mamba_v1_84842783965577
// MI455X (gfx1250) — compile-verified, matrix operand order checked by reading, not run on hardware
//
#include <hip/hip_runtime.h>
#include <hip/hip_bf16.h>
#include <math.h>

// ---- problem constants (from reference) ----
#define H_    32
#define DS_   64
#define DC_   4
#define DI_   64
#define BSZ_  4
#define L_    6
#define P_    2048
#define CIN_  45
#define BSEQ_ (BSZ_*P_)      // 8192 sequences
#define NTOK_ (BSEQ_*L_)     // 49152 tokens

// ---- mamba fused-kernel tiling ----
#define STILE    4           // sequences per workgroup
#define TOK      24          // real tokens per workgroup (STILE*L_)
#define TOKP     32          // padded to 2 WMMA M-tiles
#define XCOLS    144         // x_dbl cols padded 130 -> 9 N-tiles
#define XZC      128         // in_proj output cols (x | z)
#define MTHREADS 128         // 4 wave32
#define NWAVES   4

#define NPROG (BSZ_*P_*7)    // 57344
#define NDIAG (BSZ_*P_*3)    // 24576

typedef __attribute__((ext_vector_type(16))) _Float16 v16h;
typedef __attribute__((ext_vector_type(8)))  float    v8f;

// ---------------------------------------------------------------------------
// WMMA fragment helpers (CDNA5 wave32 16-bit layouts, CDNA5 ISA):
//  A 16x32: lanes 0-15 row M=lane, elems 0..7 = K0..7,  8..15 = K16..23
//           lanes 16-31 row M=lane-16, elems 0..7 = K8..15, 8..15 = K24..31
//  B 32x16: same striping with "row" -> column N (B consumed column-major,
//           which matches W[e][d] row-major since GEMM K = d).
//  C/D f32: VGPR r -> M = r + (lane>=16 ? 8:0), N = lane&15.
// ---------------------------------------------------------------------------
__device__ inline v16h frag_a_lds(const _Float16* base, int stride, int row,
                                  int kofs, int lane) {
  const _Float16* r0 = base + row * stride + kofs + ((lane & 16) ? 8 : 0);
  v16h a;
#pragma unroll
  for (int j = 0; j < 8; ++j) { a[j] = r0[j]; a[8 + j] = r0[16 + j]; }
  return a;
}

// B fragment from row-major f32 weights, col guaranteed in range (no branch).
__device__ inline v16h frag_b_gf32(const float* W, int ldw, int col, int kofs,
                                   int lane) {
  const float* r0 = W + (size_t)col * ldw + kofs + ((lane & 16) ? 8 : 0);
  v16h b;
#pragma unroll
  for (int j = 0; j < 8; ++j) {
    b[j]     = (_Float16)r0[j];
    b[8 + j] = (_Float16)r0[16 + j];
  }
  return b;
}

// Clamped variant for the padded N-tile of x_proj (cols 130..143 -> 0).
__device__ inline v16h frag_b_gf32_clamped(const float* W, int ldw, int col,
                                           int kofs, int lane, int ncols) {
  v16h b;
  if (col < ncols) {
    b = frag_b_gf32(W, ldw, col, kofs, lane);
  } else {
#pragma unroll
    for (int j = 0; j < 16; ++j) b[j] = (_Float16)0.f;
  }
  return b;
}

__device__ inline float silu_f(float x) { return x / (1.f + expf(-x)); }

// ---------------------------------------------------------------------------
// Kernel 1: combined-feature build + fc1 + ReLU.  thread = (token, h-lane).
// ---------------------------------------------------------------------------
__global__ __launch_bounds__(256)
void fc1_kernel(const float* __restrict__ xs, const float* __restrict__ xd,
                const float* __restrict__ xp, const float* __restrict__ xt,
                const float* __restrict__ w,  const float* __restrict__ b,
                float* __restrict__ x0) {
  int gid = blockIdx.x * blockDim.x + threadIdx.x;
  int token = gid >> 5;
  int h = gid & 31;
  if (token >= NTOK_) return;
  int seq = token / L_, l = token % L_;
  int bb = seq / P_, p = seq % P_;
  size_t base = ((size_t)(bb * L_ + l) * P_ + p);
  const float* wr = w + h * CIN_;
  float acc = b[h];
#pragma unroll
  for (int c = 0; c < 22; ++c) acc += wr[c] * xs[base * 22 + c];
#pragma unroll
  for (int c = 0; c < 12; ++c) acc += wr[22 + c] * xd[base * 12 + c];
#pragma unroll
  for (int c = 0; c < 7; ++c)  acc += wr[34 + c] * xp[base * 7 + c];
#pragma unroll
  for (int c = 0; c < 4; ++c)  acc += wr[41 + c] * xt[(size_t)(bb * L_ + l) * 4 + c];
  x0[(size_t)token * H_ + h] = fmaxf(acc, 0.f);
}

// ---------------------------------------------------------------------------
// Kernel 2: one full Mamba block (LN -> in_proj -> conv/silu -> x_proj ->
// dt/softplus -> selective scan -> gate -> out_proj -> +residual), fused.
// 4 sequences per workgroup, all intermediates in LDS, GEMMs via WMMA f16.
// ---------------------------------------------------------------------------
struct MambaParams {
  const float* ln_g;  const float* ln_b;
  const float* in_w;   // [2*DI][H]
  const float* conv_w; // [DI][DC]
  const float* conv_b; // [DI]
  const float* x_w;    // [DTR+2*DS][DI] = [130][64]
  const float* dt_w;   // [DI][2]
  const float* dt_b;   // [DI]
  const float* A_log;  // [DI][DS]
  const float* Dp;     // [DI]
  const float* out_w;  // [H][DI]
};

__global__ __launch_bounds__(MTHREADS)
void mamba_kernel(const float* __restrict__ xin, float* __restrict__ xout,
                  MambaParams pr) {
  __shared__ _Float16 s_u[TOKP * H_];       // LN'd input, f16 (WMMA A)   2KB
  __shared__ float    s_xz[TOKP * XZC];     // in_proj out [x | z];      16KB
                                            //  x-half reused as dt after conv
  __shared__ float    s_xcf[TOKP * DI_];    // conv+silu, f32             8KB
  __shared__ _Float16 s_xch[TOKP * DI_];    // conv f16; reused as y f16  4KB
  __shared__ float    s_xdbl[TOKP * XCOLS]; // [dt_raw(2)|B(64)|C(64)]   18KB

  const int tid  = threadIdx.x;
  const int lane = tid & 31;
  const int wave = tid >> 5;
  const int tok0 = blockIdx.x * TOK;        // global token base

  // --- phase 0: LayerNorm per token (lane = channel), pad rows to zero ---
  for (int tk = wave; tk < TOKP; tk += NWAVES) {
    _Float16 uv = (_Float16)0.f;
    if (tk < TOK) {
      float x = xin[(size_t)(tok0 + tk) * H_ + lane];
      float m = x;
#pragma unroll
      for (int o = 16; o > 0; o >>= 1) m += __shfl_xor(m, o);
      m *= (1.f / 32.f);
      float dx = x - m;
      float v = dx * dx;
#pragma unroll
      for (int o = 16; o > 0; o >>= 1) v += __shfl_xor(v, o);
      v *= (1.f / 32.f);
      float u = dx * rsqrtf(v + 1e-5f) * pr.ln_g[lane] + pr.ln_b[lane];
      uv = (_Float16)u;
    }
    s_u[tk * H_ + lane] = uv;
  }
  __syncthreads();

  // --- phase 1: in_proj [32 tok x 32] x [32 x 128] -> s_xz  (16 WMMA) ---
  for (int combo = wave; combo < 16; combo += NWAVES) {
    int mt = combo >> 3, nt = combo & 7;
    v16h a = frag_a_lds(s_u, H_, mt * 16 + (lane & 15), 0, lane);
    v16h b = frag_b_gf32(pr.in_w, H_, nt * 16 + (lane & 15), 0, lane);
    v8f c = {};
    c = __builtin_amdgcn_wmma_f32_16x16x32_f16(false, a, false, b, (short)0, c,
                                               false, false);
    int e = nt * 16 + (lane & 15);
#pragma unroll
    for (int r = 0; r < 8; ++r) {
      int row = mt * 16 + r + ((lane & 16) >> 1);
      s_xz[row * XZC + e] = c[r];               // branchless store
    }
  }
  __syncthreads();

  // --- phase 2: causal depthwise conv (DC=4) + SiLU -> xc (f32 + f16) ---
  for (int i = tid; i < TOKP * DI_; i += MTHREADS) {
    int tk = i >> 6, d = i & 63;
    float v = 0.f;
    if (tk < TOK) {
      int s = tk / L_, l = tk % L_;
      float acc = pr.conv_b[d];
#pragma unroll
      for (int k = 0; k < DC_; ++k) {
        int ll = l + k - (DC_ - 1);
        if (ll >= 0) acc += s_xz[(s * L_ + ll) * XZC + d] * pr.conv_w[d * DC_ + k];
      }
      v = silu_f(acc);
    }
    s_xcf[i] = v;
    s_xch[i] = (_Float16)v;
  }
  __syncthreads();

  // --- phase 3: x_proj  [32 tok x 64] x [64 x 130] -> x_dbl  (36 WMMA) ---
  for (int combo = wave; combo < 18; combo += NWAVES) {
    int mt = combo / 9, nt = combo % 9;
    v8f c = {};
#pragma unroll
    for (int kt = 0; kt < 2; ++kt) {
      v16h a = frag_a_lds(s_xch, DI_, mt * 16 + (lane & 15), kt * 32, lane);
      v16h b = frag_b_gf32_clamped(pr.x_w, DI_, nt * 16 + (lane & 15), kt * 32,
                                   lane, 130);
      c = __builtin_amdgcn_wmma_f32_16x16x32_f16(false, a, false, b, (short)0, c,
                                                 false, false);
    }
    int col = nt * 16 + (lane & 15);
#pragma unroll
    for (int r = 0; r < 8; ++r) {
      int row = mt * 16 + r + ((lane & 16) >> 1);
      s_xdbl[row * XCOLS + col] = c[r];
    }
  }
  __syncthreads();

  // --- phase 4: dt = softplus(dt_raw @ dt_w^T + dt_b) -> x-half of s_xz ---
  for (int i = tid; i < TOK * DI_; i += MTHREADS) {
    int tk = i / DI_, d = i % DI_;
    float v = s_xdbl[tk * XCOLS + 0] * pr.dt_w[d * 2 + 0] +
              s_xdbl[tk * XCOLS + 1] * pr.dt_w[d * 2 + 1] + pr.dt_b[d];
    s_xz[tk * XZC + d] = (v > 20.f) ? v : log1pf(expf(v));
  }
  __syncthreads();

  // --- phase 5: selective scan, parallel over (seq,d), n-inner, L=6 ---
  for (int it = tid; it < STILE * DI_; it += MTHREADS) {
    int s = it >> 6, d = it & 63;
    float dtv[L_], dtx[L_], y[L_];
#pragma unroll
    for (int l = 0; l < L_; ++l) {
      int tk = s * L_ + l;
      dtv[l] = s_xz[tk * XZC + d];              // dt (x-half reuse)
      dtx[l] = dtv[l] * s_xcf[tk * DI_ + d];
      y[l] = 0.f;
    }
    const float* Arow = pr.A_log + d * DS_;
    for (int n = 0; n < DS_; ++n) {
      float a = -expf(Arow[n]);
      float hs = 0.f;
#pragma unroll
      for (int l = 0; l < L_; ++l) {
        int tk = s * L_ + l;
        hs = hs * expf(dtv[l] * a) + dtx[l] * s_xdbl[tk * XCOLS + 2 + n];
        y[l] += hs * s_xdbl[tk * XCOLS + 66 + n];
      }
    }
    float Dd = pr.Dp[d];
#pragma unroll
    for (int l = 0; l < L_; ++l) {
      int tk = s * L_ + l;
      float zv = s_xz[tk * XZC + DI_ + d];      // z-half (gate)
      float g = (y[l] + s_xcf[tk * DI_ + d] * Dd) * silu_f(zv);
      s_xch[tk * DI_ + d] = (_Float16)g;        // pad rows stay 0 from phase 2
    }
  }
  __syncthreads();

  // --- phase 6: out_proj [32 tok x 64] x [64 x 32] + residual (8 WMMA) ---
  {
    int combo = wave;  // exactly 4 combos, one per wave (uniform EXEC)
    int mt = combo >> 1, nt = combo & 1;
    v8f c = {};
#pragma unroll
    for (int kt = 0; kt < 2; ++kt) {
      v16h a = frag_a_lds(s_xch, DI_, mt * 16 + (lane & 15), kt * 32, lane);
      v16h b = frag_b_gf32(pr.out_w, DI_, nt * 16 + (lane & 15), kt * 32, lane);
      c = __builtin_amdgcn_wmma_f32_16x16x32_f16(false, a, false, b, (short)0, c,
                                                 false, false);
    }
    int o = nt * 16 + (lane & 15);
#pragma unroll
    for (int r = 0; r < 8; ++r) {
      int row = mt * 16 + r + ((lane & 16) >> 1);
      if (row < TOK) {
        size_t idx = (size_t)(tok0 + row) * H_ + o;
        xout[idx] = c[r] + xin[idx];
      }
    }
  }
}

// ---------------------------------------------------------------------------
// Kernel 3: heads on last timestep: fc2+LeakyReLU, fc3 -> prog, fc4 -> diag.
// One wave per sequence; 32x32 via lane shuffles.
// ---------------------------------------------------------------------------
__global__ __launch_bounds__(256)
void heads_kernel(const float* __restrict__ x2,
                  const float* __restrict__ w2, const float* __restrict__ b2,
                  const float* __restrict__ w3, const float* __restrict__ b3,
                  const float* __restrict__ w4, const float* __restrict__ b4,
                  float* __restrict__ out) {
  int seq  = (blockIdx.x * blockDim.x + threadIdx.x) >> 5;
  int lane = threadIdx.x & 31;
  if (seq >= BSEQ_) return;
  float xl = x2[((size_t)seq * L_ + (L_ - 1)) * H_ + lane];
  float acc = b2[lane];
#pragma unroll
  for (int c = 0; c < 32; ++c) acc += w2[lane * 32 + c] * __shfl(xl, c);
  float a = (acc >= 0.f) ? acc : 0.2f * acc;  // LeakyReLU(0.2)
  float p3 = (lane < 7) ? b3[lane] : 0.f;
  float p4 = (lane < 3) ? b4[lane] : 0.f;
#pragma unroll
  for (int c = 0; c < 32; ++c) {
    float av = __shfl(a, c);
    if (lane < 7) p3 += w3[lane * 32 + c] * av;
    if (lane < 3) p4 += w4[lane * 32 + c] * av;
  }
  if (lane < 7) out[(size_t)seq * 7 + lane] = p3;
  if (lane < 3) out[NPROG + (size_t)seq * 3 + lane] = p4;
}

// ---------------------------------------------------------------------------
// Kernel 4: deterministic single-block MSE reductions (no float atomics).
// ---------------------------------------------------------------------------
__global__ __launch_bounds__(256)
void loss_kernel(float* __restrict__ out, const float* __restrict__ tp,
                 const float* __restrict__ td) {
  __shared__ float red[256];
  int tid = threadIdx.x;
  float s1 = 0.f;
  for (int i = tid; i < NPROG; i += 256) { float d = out[i] - tp[i]; s1 += d * d; }
  float s2 = 0.f;
  for (int i = tid; i < NDIAG; i += 256) { float d = out[NPROG + i] - td[i]; s2 += d * d; }
  red[tid] = s1; __syncthreads();
  for (int o = 128; o > 0; o >>= 1) { if (tid < o) red[tid] += red[tid + o]; __syncthreads(); }
  float mprog = red[0] / (float)NPROG;
  __syncthreads();
  red[tid] = s2; __syncthreads();
  for (int o = 128; o > 0; o >>= 1) { if (tid < o) red[tid] += red[tid + o]; __syncthreads(); }
  if (tid == 0) {
    out[NPROG + NDIAG + 0] = mprog;
    out[NPROG + NDIAG + 1] = red[0] / (float)NDIAG;
  }
}

// ---------------------------------------------------------------------------
extern "C" void kernel_launch(void* const* d_in, const int* in_sizes, int n_in,
                              void* d_out, int out_size, void* d_ws,
                              size_t ws_size, hipStream_t stream) {
  (void)in_sizes; (void)n_in; (void)out_size; (void)ws_size;
  const float* x_static   = (const float*)d_in[0];
  const float* x_dynamic  = (const float*)d_in[1];
  const float* x_prog     = (const float*)d_in[2];
  const float* x_time     = (const float*)d_in[3];
  const float* x_prog_inc = (const float*)d_in[4];
  const float* x_diag     = (const float*)d_in[5];
  const float* fc1_w = (const float*)d_in[6];
  const float* fc1_b = (const float*)d_in[7];
  MambaParams m1{(const float*)d_in[8],  (const float*)d_in[9],
                 (const float*)d_in[10], (const float*)d_in[11],
                 (const float*)d_in[12], (const float*)d_in[13],
                 (const float*)d_in[14], (const float*)d_in[15],
                 (const float*)d_in[16], (const float*)d_in[17],
                 (const float*)d_in[18]};
  MambaParams m2{(const float*)d_in[19], (const float*)d_in[20],
                 (const float*)d_in[21], (const float*)d_in[22],
                 (const float*)d_in[23], (const float*)d_in[24],
                 (const float*)d_in[25], (const float*)d_in[26],
                 (const float*)d_in[27], (const float*)d_in[28],
                 (const float*)d_in[29]};
  const float* fc2_w = (const float*)d_in[30];
  const float* fc2_b = (const float*)d_in[31];
  const float* fc3_w = (const float*)d_in[32];
  const float* fc3_b = (const float*)d_in[33];
  const float* fc4_w = (const float*)d_in[34];
  const float* fc4_b = (const float*)d_in[35];

  float* x0  = (float*)d_ws;                 // [NTOK, H] ping
  float* x1  = x0 + (size_t)NTOK_ * H_;      // [NTOK, H] pong
  float* out = (float*)d_out;

  fc1_kernel<<<(NTOK_ * 32) / 256, 256, 0, stream>>>(
      x_static, x_dynamic, x_prog, x_time, fc1_w, fc1_b, x0);
  mamba_kernel<<<BSEQ_ / STILE, MTHREADS, 0, stream>>>(x0, x1, m1);
  mamba_kernel<<<BSEQ_ / STILE, MTHREADS, 0, stream>>>(x1, x0, m2);
  heads_kernel<<<BSEQ_ / 8, 256, 0, stream>>>(x0, fc2_w, fc2_b, fc3_w, fc3_b,
                                              fc4_w, fc4_b, out);
  loss_kernel<<<1, 256, 0, stream>>>(out, x_prog_inc, x_diag);
}
